// MXLEmbeddings_8753143349474
// MI455X (gfx1250) — matrix operand order repaired and checked by reading, not compiled, not run on hardware
//
#include <hip/hip_runtime.h>

// ---------------- problem constants ----------------
#define EDIM 1024
#define NSEQ 4096
#define MTOT 32768          // B*N tokens
#define KP   576            // padded K (18 tiles of 32); logical K = 558
#define KT   18             // K tiles of 32
#define MT   2048           // 16-row M tiles (pack granularity)
#define MT2  1024           // 32-row M tiles (GEMM block granularity)
#define CT   64             // 16-col tiles (EDIM/16)
#define ATILE_HALF (KT * 32 * 16)         // halves per 16-row packed tile (9216)
#define ATILE2_B   (2 * ATILE_HALF * 2)   // bytes per 32-row block tile (36864)

typedef __attribute__((ext_vector_type(16))) _Float16 v16h;
typedef __attribute__((ext_vector_type(8)))  float    v8f;
typedef __attribute__((ext_vector_type(4)))  float    v4f;
typedef __attribute__((ext_vector_type(4)))  unsigned int v4u;
typedef __attribute__((ext_vector_type(8)))  int      v8i;
typedef __attribute__((ext_vector_type(4)))  int      v4i;

union Frag {
    v16h     h16;
    v4f      f4[2];
    _Float16 h[16];
};

struct Ptrs13 { const float* p[13]; };

__device__ __forceinline__ int kbase(int i) {
    const int b[13] = {0,193,195,388,516,522,538,540,542,544,552,555,557};
    return b[i];
}

#if defined(__HIP_DEVICE_COMPILE__) && \
    __has_builtin(__builtin_amdgcn_tensor_load_to_lds) && \
    __has_builtin(__builtin_amdgcn_s_wait_tensorcnt)
#define USE_TDM 1
#else
#define USE_TDM 0
#endif

// ------------------------------------------------------------------
// Stage 1: A in WMMA fragment order (f16), fusing the shift.
// A as packed here (contiguous k, not ISA table order): lane L, half h (v=h/2,p=h&1):
//   M = L&15, grp = L>>4, K = grp*16 + h
// ------------------------------------------------------------------
__global__ __launch_bounds__(256)
void pack_a_kernel(Ptrs13 in, _Float16* __restrict__ Af) {
    int gid  = blockIdx.x * 256 + threadIdx.x;      // < MT*KT*32
    int lane = gid & 31;
    int kt   = (gid >> 5) % KT;
    int mt   = gid / (KT * 32);
    int grp  = lane >> 4;
    int m    = mt * 16 + (lane & 15);
    bool valid = (m & (NSEQ - 1)) != 0;             // shifted row 0 is zero
    int srow = m - 1;
    Frag f;
#pragma unroll
    for (int h = 0; h < 16; ++h) {
        int v = h >> 1;
        int k = kt * 32 + (grp << 4) + h;
        float val = 0.0f;
        if (valid) {
            if (k < 557) {
                int s = 0;
#pragma unroll
                for (int i = 1; i < 12; ++i) if (k >= kbase(i)) s = i;
                int vs = kbase(s + 1) - kbase(s);
                val = in.p[s][srow * vs + (k - kbase(s))];
            } else if (k == 557) {
                val = in.p[12][srow];               // pad column
            }
        }
        f.h[h] = (_Float16)val;
    }
    v4f* o = (v4f*)(Af + (size_t)gid * 16);
    o[0] = f.f4[0];
    o[1] = f.f4[1];
}

// ------------------------------------------------------------------
// Stage 2a: B in fragment order (32x16): lane L: N=L&15, grp=L>>4,
// half h: K = grp*16 + h.  W_<s> is [E,V] row-major; W_mask is [E,1].
// ------------------------------------------------------------------
__global__ __launch_bounds__(256)
void pack_b_kernel(Ptrs13 w, _Float16* __restrict__ Bf) {
    int gid  = blockIdx.x * 256 + threadIdx.x;      // < KT*CT*32
    int lane = gid & 31;
    int ct   = (gid >> 5) % CT;
    int kt   = gid / (CT * 32);
    int e    = ct * 16 + (lane & 15);
    int kb   = kt * 32 + ((lane >> 4) << 4);
    Frag f;
#pragma unroll
    for (int h = 0; h < 16; ++h) {
        int k = kb + h;
        float val = 0.0f;
        if (k < 557) {
            int s = 0;
#pragma unroll
            for (int i = 1; i < 12; ++i) if (k >= kbase(i)) s = i;
            int vs = kbase(s + 1) - kbase(s);
            val = w.p[s][e * vs + (k - kbase(s))];
        } else if (k == 557) {
            val = w.p[12][e];
        }
        f.h[h] = (_Float16)val;
    }
    v4f* o = (v4f*)(Bf + (size_t)gid * 16);
    o[0] = f.f4[0];
    o[1] = f.f4[1];
}

// Stage 2b: fold all 13 biases into one vector.
__global__ __launch_bounds__(256)
void bias_kernel(Ptrs13 b, float* __restrict__ bias) {
    int e = blockIdx.x * 256 + threadIdx.x;
    if (e < EDIM) {
        float s = 0.0f;
#pragma unroll
        for (int i = 0; i < 13; ++i) s += b.p[i][e];
        bias[e] = s;
    }
}

// ------------------------------------------------------------------
// Stage 3: GEMM (f16 WMMA, fp32 acc) + bias + fused LayerNorm.
// Block = 256 threads (8 waves) owns 32 tokens x full E=1024.
// Wave w: rows 0..31 (two 16-row tiles), cols [w*128, w*128+128).
// A tile (36 KB) staged via TDM; B fragments in a 3-deep ring with
// +2 lookahead so two L2 loads are always in flight per wave.
// ------------------------------------------------------------------
struct __align__(16) SMem {
    _Float16 As[2 * ATILE_HALF];   // 36 KB — must be first (TDM target)
    float red1[32][8];
    float red2[32][8];
    float smu[32];
    float srs[32];
};

__device__ __forceinline__ void loadB(Frag& f, const _Float16* p) {
    const v4f* q = (const v4f*)p;
    f.f4[0] = q[0];
    f.f4[1] = q[1];
}

__global__ __launch_bounds__(256)
void gemm_ln_kernel(const _Float16* __restrict__ Af,
                    const _Float16* __restrict__ Bf,
                    const float* __restrict__ bias,
                    const float* __restrict__ gamma,
                    const float* __restrict__ beta,
                    float* __restrict__ out) {
    __shared__ SMem sm;

    const int tid  = threadIdx.x;
    const int wave = tid >> 5;
    const int lane = tid & 31;
    const int grp  = lane >> 4;
    const int lid  = lane & 15;
    const int mt2  = blockIdx.x;

#if USE_TDM
    if (wave == 0) {
        // D# for a 1-D tile: 4608 x 8-byte elements = 36864 B, global -> LDS
        unsigned lds = (unsigned)(unsigned long long)(void*)sm.As;
        unsigned long long ga =
            (unsigned long long)(const void*)(Af + (size_t)mt2 * (2 * ATILE_HALF));
        const unsigned nelem = ATILE2_B / 8;        // 4608
        v4u g0 = { 1u,                               // count=1, normal mode
                   lds,                              // lds_addr
                   (unsigned)ga,                     // global_addr[31:0]
                   (unsigned)((ga >> 32) & 0x01FFFFFFu) | 0x80000000u }; // +type=2
        v8i g1 = { (int)0x00030000,                  // data_size=3 (8B)
                   (int)((nelem & 0xFFFFu) << 16),   // tensor_dim0[15:0]
                   (int)(((nelem >> 16) & 0xFFFFu) | (1u << 16)), // dim0 hi | tensor_dim1=1
                   (int)(nelem << 16),               // tile_dim0
                   0,                                // tile_dim1=0, tile_dim2=0
                   (int)nelem,                       // tensor_dim0_stride lo
                   0, 0 };
        v4i gz4 = { 0, 0, 0, 0 };
        v8i gz8 = { 0, 0, 0, 0, 0, 0, 0, 0 };
        __builtin_amdgcn_tensor_load_to_lds(g0, g1, gz4, gz4, gz8, 0);
        __builtin_amdgcn_s_wait_tensorcnt(0);
    }
#else
    {
        const v4f* src = (const v4f*)(Af + (size_t)mt2 * (2 * ATILE_HALF));
        v4f* dst = (v4f*)sm.As;
        for (int i = tid; i < ATILE2_B / 16; i += 256) dst[i] = src[i];
    }
#endif
    __syncthreads();

    v8f acc[2][8];
#pragma unroll
    for (int t = 0; t < 2; ++t)
#pragma unroll
        for (int ns = 0; ns < 8; ++ns) acc[t][ns] = {};

    const int ct0 = wave << 3;
    auto baddr = [&](int idx) {
        int kt2 = idx >> 3, ns2 = idx & 7;
        return Bf + (size_t)(((kt2 * CT + ct0 + ns2) * 32 + lane) << 4);
    };

    Frag b[3];
    loadB(b[0], baddr(0));
    loadB(b[1], baddr(1));

#pragma unroll
    for (int kt = 0; kt < KT; ++kt) {
        Frag a0, a1;
        {
            const v4f* p0 = (const v4f*)(sm.As + ((kt * 32 + lane) << 4));
            const v4f* p1 = (const v4f*)(sm.As + ATILE_HALF + ((kt * 32 + lane) << 4));
            a0.f4[0] = p0[0]; a0.f4[1] = p0[1];
            a1.f4[0] = p1[0]; a1.f4[1] = p1[1];
        }
#pragma unroll
        for (int ns = 0; ns < 8; ++ns) {
            const int idx = kt * 8 + ns;
            if (idx + 2 < KT * 8) loadB(b[(idx + 2) % 3], baddr(idx + 2));
            const Frag& bc = b[idx % 3];
            acc[0][ns] = __builtin_amdgcn_wmma_f32_16x16x32_f16(
                false, a0.h16, false, bc.h16, (short)0, acc[0][ns], false, false);
            acc[1][ns] = __builtin_amdgcn_wmma_f32_16x16x32_f16(
                false, a1.h16, false, bc.h16, (short)0, acc[1][ns], false, false);
        }
    }

    // bias add
#pragma unroll
    for (int ns = 0; ns < 8; ++ns) {
        int col = ((ct0 + ns) << 4) + lid;
        float bb = bias[col];
#pragma unroll
        for (int t = 0; t < 2; ++t)
#pragma unroll
            for (int r = 0; r < 8; ++r) acc[t][ns][r] += bb;
    }

    // per-row partial sums + butterfly reduce over the 16 lanes of each half
#pragma unroll
    for (int t = 0; t < 2; ++t) {
#pragma unroll
        for (int r = 0; r < 8; ++r) {
            float s1 = 0.0f, s2 = 0.0f;
#pragma unroll
            for (int ns = 0; ns < 8; ++ns) {
                float x = acc[t][ns][r];
                s1 += x;
                s2 += x * x;
            }
#pragma unroll
            for (int off = 8; off >= 1; off >>= 1) {
                s1 += __shfl_xor(s1, off, 32);
                s2 += __shfl_xor(s2, off, 32);
            }
            if (lid == 0) {
                int row = t * 16 + r + (grp << 3);
                sm.red1[row][wave] = s1;
                sm.red2[row][wave] = s2;
            }
        }
    }
    __syncthreads();
    if (tid < 32) {
        float a = 0.0f, b2 = 0.0f;
#pragma unroll
        for (int w = 0; w < 8; ++w) { a += sm.red1[tid][w]; b2 += sm.red2[tid][w]; }
        float mu  = a * (1.0f / EDIM);
        float var = b2 * (1.0f / EDIM) - mu * mu;
        sm.smu[tid] = mu;
        sm.srs[tid] = rsqrtf(var + 1e-12f);
    }
    __syncthreads();

#pragma unroll
    for (int ns = 0; ns < 8; ++ns) {
        int col = ((ct0 + ns) << 4) + lid;
        float g  = gamma[col];
        float bt = beta[col];
#pragma unroll
        for (int t = 0; t < 2; ++t)
#pragma unroll
            for (int r = 0; r < 8; ++r) {
                int row = t * 16 + r + (grp << 3);
                float y = (acc[t][ns][r] - sm.smu[row]) * sm.srs[row] * g + bt;
                out[(size_t)(mt2 * 32 + row) * EDIM + col] = y;
            }
    }
}

// ------------------------------------------------------------------
// Inputs: 0..11 streams, 12 pad, (W_s,b_s) pairs 13..36,
// W_mask=37, b_mask=38, ln_gamma=39, ln_beta=40.
// ------------------------------------------------------------------
extern "C" void kernel_launch(void* const* d_in, const int* in_sizes, int n_in,
                              void* d_out, int out_size, void* d_ws, size_t ws_size,
                              hipStream_t stream) {
    (void)in_sizes; (void)n_in; (void)out_size; (void)ws_size;

    Ptrs13 A;
    for (int i = 0; i < 12; ++i) A.p[i] = (const float*)d_in[i];
    A.p[12] = (const float*)d_in[12];                 // pad

    Ptrs13 W, Bb;
    for (int i = 0; i < 12; ++i) {
        W.p[i]  = (const float*)d_in[13 + 2 * i];
        Bb.p[i] = (const float*)d_in[14 + 2 * i];
    }
    W.p[12]  = (const float*)d_in[37];                // W_mask
    Bb.p[12] = (const float*)d_in[38];                // b_mask
    const float* gamma = (const float*)d_in[39];
    const float* beta  = (const float*)d_in[40];

    _Float16* Af = (_Float16*)d_ws;                                   // 37.75 MB
    _Float16* Bf = (_Float16*)((char*)d_ws + (size_t)MTOT * KP * 2);  // 1.18 MB
    float*  bias = (float*)((char*)d_ws + (size_t)MTOT * KP * 2
                                        + (size_t)KP * EDIM * 2);     // 4 KB

    pack_a_kernel<<<(MT * KT * 32) / 256, 256, 0, stream>>>(A, Af);
    pack_b_kernel<<<(KT * CT * 32) / 256, 256, 0, stream>>>(W, Bf);
    bias_kernel<<<EDIM / 256, 256, 0, stream>>>(Bb, bias);
    gemm_ln_kernel<<<MT2, 256, 0, stream>>>(Af, Bf, bias, gamma, beta, (float*)d_out);
}
